// Attention_2_87393994539659
// MI455X (gfx1250) — matrix operand order repaired and checked by reading, not compiled, not run on hardware
//
#include <hip/hip_runtime.h>

// ---------------------------------------------------------------------------
// SAGAN attention block for gfx1250 (MI455X), bf16 WMMA + fp32 accumulate.
//   B=8, H=W=64, C=256, N=4096 queries/batch, M=1024 keys/batch
// All WMMA fragments are fed by contiguous 32-byte-per-lane b128 loads:
//   - weights pre-packed to B-fragment layout (pack_weights_kernel)
//   - pooled h written in B-fragment layout by the projection kernel
//   - softmax probabilities packed to A-fragment bf16 layout in LDS
// Hot loops are software-pipelined (loads for step k+1 issued before the
// WMMAs of step k) so VMEM latency overlaps the matrix pipe.
// ---------------------------------------------------------------------------

typedef __attribute__((ext_vector_type(16))) __bf16 bf16x16;
typedef __attribute__((ext_vector_type(8)))  __bf16 bf16x8;
typedef __attribute__((ext_vector_type(8)))  float  f32x8;

#define DEV __device__ __forceinline__

DEV f32x8 wmma_bf16(bf16x16 a, bf16x16 b, f32x8 c) {
  return __builtin_amdgcn_wmma_f32_16x16x32_bf16(
      /*neg_a=*/false, a, /*neg_b=*/false, b,
      /*c_mod=*/(short)0, c, /*reuse_a=*/false, /*reuse_b=*/false);
}

// Load a fragment stored contiguously: lane l owns 16 bf16 at p + l*16.
DEV bf16x16 ld_frag(const __bf16* p) {
  return *(const bf16x16*)(p + (size_t)(threadIdx.x & 31) * 16);
}

// Packed B-fragment array: fragment (kc, ct) at offset (kc*NT + ct)*512.
DEV bf16x16 b_frag_packed(const __bf16* P, int NT, int kc, int ct) {
  return ld_frag(P + (size_t)(kc * NT + ct) * 512);
}

// A fragment (16x32, MxK) from row-major f32 source (global or LDS).
// Lanes 0-15: M=lane, K={0..15}; lanes 16-31: M=lane-16, K={16..31}.
DEV bf16x16 a_frag_f32(const float* base, int ld, int k0) {
  const int lane = threadIdx.x & 31;
  const int m  = lane & 15;
  const int kb = (lane >> 4) << 4;
  const float* p = base + (size_t)m * ld + k0 + kb;
  const f32x8 lo = *(const f32x8*)p;
  const f32x8 hi = *(const f32x8*)(p + 8);
  bf16x16 v;
#pragma unroll
  for (int i = 0; i < 8; ++i) { v[i] = (__bf16)lo[i]; v[i + 8] = (__bf16)hi[i]; }
  return v;
}

// Same A layout, bf16 row-major source (two 16-byte runs per lane).
DEV bf16x16 a_frag_bf(const __bf16* base, int ld, int k0) {
  const int lane = threadIdx.x & 31;
  const int m  = lane & 15;
  const int kb = (lane >> 4) << 4;
  const __bf16* p = base + (size_t)m * ld + k0 + kb;
  const bf16x8 lo = *(const bf16x8*)p;
  const bf16x8 hi = *(const bf16x8*)(p + 8);
  bf16x16 v;
#pragma unroll
  for (int i = 0; i < 8; ++i) { v[i] = lo[i]; v[i + 8] = hi[i]; }
  return v;
}

// B fragment for S = G * F^T: B(k,n) = F[(n0+n)*32 + k]; contiguous 32B per lane.
DEV bf16x16 b_frag_T32(const __bf16* F, int n0) {
  const int lane = threadIdx.x & 31;
  const int n  = lane & 15;
  const int kh = (lane >> 4) << 4;
  return *(const bf16x16*)(F + (size_t)(n0 + n) * 32 + kh);
}

// C/D tile (16x16 f32) scatter into row-major f32 LDS region.
// VGPR r holds (M = r + 8*(lane>=16), N = lane&15).
DEV void c_store_lds(float* L, int ld, f32x8 c) {
  const int lane = threadIdx.x & 31;
  const int n  = lane & 15;
  const int mb = (lane >> 4) << 3;
#pragma unroll
  for (int r = 0; r < 8; ++r) L[(size_t)(mb + r) * ld + n] = c[r];
}

// ---------------------------------------------------------------------------
// Kernel 0: pack fp32 weights into bf16 B-fragment layout.
// Packed idx = ((kc*NT + ct)*32 + lane)*16 + i  maps to  W[kc*32 + kh + i][ct*16 + n]
// with kh = (lane>>4)*16, n = lane&15.
// ---------------------------------------------------------------------------
DEV void pack_one(const float* W, int N, __bf16* dst, int idx) {
  const int NT   = N >> 4;
  const int i    = idx & 15;
  const int lane = (idx >> 4) & 31;
  const int rest = idx >> 9;
  const int ct   = rest % NT;
  const int kc   = rest / NT;
  const int kh   = (lane >> 4) << 4;
  const int n    = lane & 15;
  dst[idx] = (__bf16)W[(size_t)(kc * 32 + kh + i) * N + ct * 16 + n];
}

__global__ void __launch_bounds__(256)
pack_weights_kernel(const float* __restrict__ Wg, const float* __restrict__ Wf,
                    const float* __restrict__ Wh, const float* __restrict__ Wo,
                    __bf16* __restrict__ Wg_p, __bf16* __restrict__ Wf_p,
                    __bf16* __restrict__ Wh_p, __bf16* __restrict__ Wo_p) {
  const int idx = blockIdx.x * 256 + threadIdx.x;
  if (idx < 8192)        pack_one(Wg,  32, Wg_p, idx);            // 256x32
  else if (idx < 16384)  pack_one(Wf,  32, Wf_p, idx - 8192);     // 256x32
  else if (idx < 49152)  pack_one(Wh, 128, Wh_p, idx - 16384);    // 256x128
  else if (idx < 81920)  pack_one(Wo, 256, Wo_p, idx - 49152);    // 128x256
}

// ---------------------------------------------------------------------------
// Kernel 1: fused projections  g = x*Wg,  f = maxpool(x*Wf),  h = maxpool(x*Wh)
// One workgroup (8 waves) per (batch, pooled-row): 128 pixels = 2 image rows.
// GEMM [128x256]*[256x192] via bf16 WMMA; pool + bias in epilogue.
// LDS: 128 x 192 f32 = 96 KB (cols 0-31 g, 32-63 f_pre, 64-191 h_pre).
// ---------------------------------------------------------------------------
__global__ void __launch_bounds__(256)
sagan_proj_kernel(const float* __restrict__ x,
                  const __bf16* __restrict__ Wg_p, const __bf16* __restrict__ Wf_p,
                  const __bf16* __restrict__ Wh_p,
                  const float* __restrict__ bf_, const float* __restrict__ bg,
                  const float* __restrict__ bh,
                  __bf16* __restrict__ g_ws, __bf16* __restrict__ f_ws,
                  __bf16* __restrict__ h_pk) {
  extern __shared__ float lds[];                 // [128][192]
  const int tid  = threadIdx.x;
  const int wave = tid >> 5;
  const int b  = blockIdx.x >> 5;                // 32 pooled rows per batch
  const int iy = blockIdx.x & 31;
  const float* xb = x + (size_t)(b * 4096 + iy * 128) * 256;

  // ---- per-wave WMMA GEMM on rows [wave*16, wave*16+16) ----
  const float* arow = xb + (size_t)(wave * 16) * 256;
  bf16x16 a[8];
#pragma unroll
  for (int kc = 0; kc < 8; ++kc) a[kc] = a_frag_f32(arow, 256, kc * 32);

  for (int ct = 0; ct < 12; ++ct) {              // 2 g-tiles, 2 f-tiles, 8 h-tiles
    const __bf16* Wsel; int NT, ctw;
    if (ct < 2)      { Wsel = Wg_p; NT = 2; ctw = ct; }
    else if (ct < 4) { Wsel = Wf_p; NT = 2; ctw = ct - 2; }
    else             { Wsel = Wh_p; NT = 8; ctw = ct - 4; }
    // Group all 16 b128 loads (one clause, one wait), then 8 back-to-back WMMAs.
    bf16x16 bfr[8];
#pragma unroll
    for (int kc = 0; kc < 8; ++kc) bfr[kc] = b_frag_packed(Wsel, NT, kc, ctw);
    f32x8 acc = {};
#pragma unroll
    for (int kc = 0; kc < 8; ++kc) acc = wmma_bf16(a[kc], bfr[kc], acc);
    c_store_lds(lds + (size_t)(wave * 16) * 192 + ct * 16, 192, acc);
  }
  __syncthreads();

  // ---- g write-out (no pooling): 128 x 32 ----
  for (int idx = tid; idx < 128 * 32; idx += 256) {
    const int p = idx >> 5, c = idx & 31;
    const float v = lds[(size_t)p * 192 + c] + bg[c];
    const size_t row = (size_t)b * 4096 + iy * 128 + p;
    g_ws[row * 32 + c] = (__bf16)v;
  }
  // ---- f: 2x2 maxpool + bias: 32 x 32 (row-major, consumed as F^T B-frags) ----
  for (int idx = tid; idx < 32 * 32; idx += 256) {
    const int ox = idx >> 5, c = idx & 31;
    const int p0 = 2 * ox;
    float v = fmaxf(fmaxf(lds[(size_t)p0 * 192 + 32 + c],
                          lds[(size_t)(p0 + 1) * 192 + 32 + c]),
                    fmaxf(lds[(size_t)(p0 + 64) * 192 + 32 + c],
                          lds[(size_t)(p0 + 65) * 192 + 32 + c]));
    v += bf_[c];
    const size_t row = (size_t)b * 1024 + iy * 32 + ox;
    f_ws[row * 32 + c] = (__bf16)v;
  }
  // ---- h: 2x2 maxpool + bias: 32 x 128, written in packed B-fragment layout ----
  for (int idx = tid; idx < 32 * 128; idx += 256) {
    const int ox = idx >> 7, c = idx & 127;
    const int p0 = 2 * ox;
    float v = fmaxf(fmaxf(lds[(size_t)p0 * 192 + 64 + c],
                          lds[(size_t)(p0 + 1) * 192 + 64 + c]),
                    fmaxf(lds[(size_t)(p0 + 64) * 192 + 64 + c],
                          lds[(size_t)(p0 + 65) * 192 + 64 + c]));
    v += bh[c];
    const int m    = iy * 32 + ox;          // key row within batch (K dim)
    const int kc   = m >> 5, kk = m & 31;
    const int lane = ((kk >> 4) << 4) | (c & 15);
    const int i    = kk & 15;
    const int ctn  = c >> 4;
    const size_t off = ((size_t)(kc * 8 + ctn) * 32 + lane) * 16 + i;
    h_pk[(size_t)b * 32768 + off] = (__bf16)v;
  }
}

// ---------------------------------------------------------------------------
// Kernel 2: fused attention + output projection + residual.
// One workgroup (4 waves) per 16 query rows; score block S[16x1024] lives
// entirely in LDS — never touches HBM.
//   P1: S = G * F^T          (64 WMMA tiles, K=32)
//   P2: row softmax; normalized P packed to bf16 A-fragment LDS layout
//   P3: O = P * H            (software-pipelined, all b128 feeds)
//   P4: Out = gamma*(O*Wo + bo) + x
// LDS: 64 KB sS + 32 KB sPb + 8 KB sO = 104 KB.
// ---------------------------------------------------------------------------
__global__ void __launch_bounds__(128)
sagan_attn_kernel(const __bf16* __restrict__ g_ws, const __bf16* __restrict__ f_ws,
                  const __bf16* __restrict__ h_pk, const __bf16* __restrict__ Wo_p,
                  const float* __restrict__ bo, const float* __restrict__ gamma,
                  const float* __restrict__ x, float* __restrict__ out) {
  extern __shared__ char smem[];
  float*  sS  = (float*)smem;                    // [16][1024] scores (f32)
  __bf16* sPb = (__bf16*)(smem + 65536);         // packed P A-fragments (32 kc * 512)
  float*  sO  = (float*)(smem + 65536 + 32768);  // [16][128] attention output
  const int tid  = threadIdx.x;
  const int wave = tid >> 5;
  const size_t row0 = (size_t)blockIdx.x * 16;   // global query row
  const int b = (int)(row0 >> 12);               // 4096 rows per batch
  const __bf16* Fb = f_ws + (size_t)b * 1024 * 32;
  const __bf16* Hb = h_pk + (size_t)b * 32768;

  __builtin_prefetch(x + row0 * 256, 0, 0);      // residual tile -> global_prefetch_b8

  // ---- Phase 1: S = G[16x32] * F^T[32x1024] (pipelined over 16 m-tiles) ----
  {
    const bf16x16 ag = a_frag_bf(g_ws + row0 * 32, 32, 0);
    const int mt0 = wave * 16;
    bf16x16 bcur = b_frag_T32(Fb, mt0 * 16);
    for (int mt = mt0; mt < mt0 + 15; ++mt) {
      const bf16x16 bnxt = b_frag_T32(Fb, (mt + 1) * 16);
      f32x8 acc = {};
      acc = wmma_bf16(ag, bcur, acc);
      c_store_lds(sS + mt * 16, 1024, acc);
      bcur = bnxt;
    }
    f32x8 acc = {};
    acc = wmma_bf16(ag, bcur, acc);
    c_store_lds(sS + (mt0 + 15) * 16, 1024, acc);
  }
  __syncthreads();

  // ---- Phase 2: softmax along M=1024 (8 threads per row); pack bf16 P ----
  {
    const int r = tid >> 3, sub = tid & 7;
    float* rowp = sS + (size_t)r * 1024 + sub * 128;
    float mx = -3.4e38f;
#pragma unroll 4
    for (int c = 0; c < 128; ++c) mx = fmaxf(mx, rowp[c]);
    mx = fmaxf(mx, __shfl_xor(mx, 1));
    mx = fmaxf(mx, __shfl_xor(mx, 2));
    mx = fmaxf(mx, __shfl_xor(mx, 4));
    float sm = 0.f;
#pragma unroll 4
    for (int c = 0; c < 128; ++c) { const float e = __expf(rowp[c] - mx); rowp[c] = e; sm += e; }
    sm += __shfl_xor(sm, 1);
    sm += __shfl_xor(sm, 2);
    sm += __shfl_xor(sm, 4);
    const float inv = 1.0f / sm;
    // normalize + scatter into A-fragment bf16 layout:
    //   kk in {0..15}->lh0,i=kk; {16..31}->lh1,i=kk&15
#pragma unroll 4
    for (int c0 = 0; c0 < 128; ++c0) {
      const int c  = sub * 128 + c0;
      const int kc = c >> 5, kk = c & 31;
      const int lh = kk >> 4;
      const int i  = kk & 15;
      const int lane = (lh << 4) | r;
      sPb[(size_t)(kc * 32 + lane) * 16 + i] = (__bf16)(rowp[c0] * inv);
    }
  }
  __syncthreads();

  // ---- Phase 3: O = P[16x1024] * H[1024x128]; wave owns 32 output cols.
  //      One-deep software pipeline: loads for kc+1 issued before WMMAs of kc.
  {
    f32x8 acc0 = {}, acc1 = {};
    const int ct0 = wave * 2;
    bf16x16 ap = ld_frag(sPb);
    bf16x16 h0 = b_frag_packed(Hb, 8, 0, ct0);
    bf16x16 h1 = b_frag_packed(Hb, 8, 0, ct0 + 1);
    for (int kc = 0; kc < 31; ++kc) {
      const bf16x16 ap_n = ld_frag(sPb + (size_t)(kc + 1) * 512);
      const bf16x16 h0_n = b_frag_packed(Hb, 8, kc + 1, ct0);
      const bf16x16 h1_n = b_frag_packed(Hb, 8, kc + 1, ct0 + 1);
      acc0 = wmma_bf16(ap, h0, acc0);
      acc1 = wmma_bf16(ap, h1, acc1);
      ap = ap_n; h0 = h0_n; h1 = h1_n;
    }
    acc0 = wmma_bf16(ap, h0, acc0);
    acc1 = wmma_bf16(ap, h1, acc1);
    c_store_lds(sO + ct0 * 16,       128, acc0);
    c_store_lds(sO + (ct0 + 1) * 16, 128, acc1);
  }
  __syncthreads();

  // ---- Phase 4: Out = gamma*(O[16x128]*Wo[128x256] + bo) + x ----
  {
    f32x8 acc[4] = {};
    for (int kc = 0; kc < 4; ++kc) {
      const bf16x16 ao = a_frag_f32(sO, 128, kc * 32);
      bf16x16 bw[4];
#pragma unroll
      for (int j = 0; j < 4; ++j) bw[j] = b_frag_packed(Wo_p, 16, kc, wave * 4 + j);
#pragma unroll
      for (int j = 0; j < 4; ++j) acc[j] = wmma_bf16(ao, bw[j], acc[j]);
    }
    const float gam = gamma[0];
    const int lane = tid & 31;
    const int n = lane & 15, mb = (lane >> 4) << 3;
#pragma unroll
    for (int j = 0; j < 4; ++j) {
      const int col = (wave * 4 + j) * 16 + n;
      const float bias = bo[col];
#pragma unroll
      for (int r = 0; r < 8; ++r) {
        const size_t off = (row0 + mb + r) * 256 + col;
        out[off] = gam * (acc[j][r] + bias) + x[off];
      }
    }
  }
}

// ---------------------------------------------------------------------------
// Host launcher
// ---------------------------------------------------------------------------
extern "C" void kernel_launch(void* const* d_in, const int* in_sizes, int n_in,
                              void* d_out, int out_size, void* d_ws, size_t ws_size,
                              hipStream_t stream) {
  (void)in_sizes; (void)n_in; (void)out_size; (void)ws_size;
  const float* x     = (const float*)d_in[0];
  const float* Wf    = (const float*)d_in[1];
  const float* bf_   = (const float*)d_in[2];
  const float* Wg    = (const float*)d_in[3];
  const float* bg    = (const float*)d_in[4];
  const float* Wh    = (const float*)d_in[5];
  const float* bh    = (const float*)d_in[6];
  const float* Wo    = (const float*)d_in[7];
  const float* bo    = (const float*)d_in[8];
  const float* gamma = (const float*)d_in[9];
  float* out = (float*)d_out;

  char* ws = (char*)d_ws;
  __bf16* g_ws = (__bf16*)ws;                                   // 32768*32*2 = 2 MiB
  __bf16* f_ws = (__bf16*)(ws + (size_t)2097152);               //  8192*32*2 = 512 KiB
  __bf16* h_pk = (__bf16*)(ws + (size_t)2621440);               // 8*32768*2  = 512 KiB
  __bf16* Wg_p = (__bf16*)(ws + (size_t)3145728);               // 16 KiB
  __bf16* Wf_p = (__bf16*)(ws + (size_t)3162112);               // 16 KiB
  __bf16* Wh_p = (__bf16*)(ws + (size_t)3178496);               // 64 KiB
  __bf16* Wo_p = (__bf16*)(ws + (size_t)3244032);               // 64 KiB

  // Weight packing: 81920 packed elements.
  pack_weights_kernel<<<320, 256, 0, stream>>>(Wg, Wf, Wh, Wo, Wg_p, Wf_p, Wh_p, Wo_p);

  // Projections: B*32 = 256 workgroups, 8 waves each, 96 KB dynamic LDS.
  sagan_proj_kernel<<<256, 256, 128 * 192 * sizeof(float), stream>>>(
      x, Wg_p, Wf_p, Wh_p, bf_, bg, bh, g_ws, f_ws, h_pk);

  // Attention: 32768/16 = 2048 workgroups, 4 waves each, 104 KB dynamic LDS.
  sagan_attn_kernel<<<2048, 128, 65536 + 32768 + 8192, stream>>>(
      g_ws, f_ws, h_pk, Wo_p, bo, gamma, x, out);
}
